// TranxParser_68693706932620
// MI455X (gfx1250) — matrix operand order repaired and checked by reading, not compiled, not run on hardware
//
#include <hip/hip_runtime.h>
#include <hip/hip_bf16.h>
#include <math.h>

typedef __attribute__((ext_vector_type(16))) _Float16 v16h;
typedef __attribute__((ext_vector_type(8)))  _Float16 h8;
typedef __attribute__((ext_vector_type(4)))  _Float16 h4;
typedef __attribute__((ext_vector_type(8)))  float    v8f;

// ---------------- WMMA fragment helpers (CDNA5 gfx1250, wave32) ----------------
__device__ __forceinline__ v16h cat8(h8 a, h8 b) {
  return __builtin_shufflevector(a, b, 0, 1, 2, 3, 4, 5, 6, 7, 8, 9, 10, 11, 12, 13, 14, 15);
}
// A-fragment 16x32 f16 row-major (ld halves): lane (hi,m); elems = K k0+hi*16..k0+hi*16+15 contiguous
__device__ __forceinline__ v16h frag_a_f16(const _Float16* base, int ld, int m0, int k0, int lane) {
  const _Float16* p = base + (m0 + (lane & 15)) * ld + k0 + ((lane >> 4) & 1) * 16;
  return cat8(*(const h8*)p, *(const h8*)(p + 8));
}
// B-fragment 32x16 (KxN) from W (N x K row-major): lane (hi,n); elems = K kb..kb+15 contiguous
__device__ __forceinline__ v16h frag_b_f16(const _Float16* base, int ld, int n0, int k0, int lane) {
  const _Float16* p = base + (n0 + (lane & 15)) * ld + k0 + ((lane >> 4) & 1) * 16;
  return cat8(*(const h8*)p, *(const h8*)(p + 8));
}
__device__ __forceinline__ v8f wmma16(v16h a, v16h b, v8f c) {
  return __builtin_amdgcn_wmma_f32_16x16x32_f16(false, a, false, b, (short)0, c, false, false);
}
// C/D layout: elem r -> row = r + (lane>>4)*8 , col = lane&15
__device__ __forceinline__ float sigf(float x) { return 1.f / (1.f + expf(-x)); }

// ---------------- prep kernels ----------------
__global__ void k_tof16(const float* __restrict__ s, _Float16* __restrict__ d, int n) {
  int i = blockIdx.x * blockDim.x + threadIdx.x;
  if (i < n) d[i] = (_Float16)s[i];
}
__global__ void k_wd16(const float* __restrict__ Wih, const float* __restrict__ Whh,
                       _Float16* __restrict__ wd) {   // [Wih_d | Whh_d] -> 1024 x 960
  int i = blockIdx.x * blockDim.x + threadIdx.x;
  if (i >= 1024 * 960) return;
  int n = i / 960, k = i % 960;
  wd[i] = (_Float16)((k < 704) ? Wih[n * 704 + k] : Whh[n * 256 + (k - 704)]);
}
__global__ void k_prod16(const float* __restrict__ p, _Float16* __restrict__ d) {  // pad 201->208
  int i = blockIdx.x * blockDim.x + threadIdx.x;
  if (i >= 208 * 128) return;
  int n = i >> 7;
  d[i] = (n < 201) ? (_Float16)p[i] : (_Float16)0.f;
}
__global__ void k_embed(const int* __restrict__ tok, const float* __restrict__ se,
                        _Float16* __restrict__ emb16) {
  int i = blockIdx.x * blockDim.x + threadIdx.x;
  if (i >= 64 * 100 * 128) return;
  int e = i & 127, bs = i >> 7;
  emb16[i] = (_Float16)se[tok[bs] * 128 + e];
}

// ------- generic C = A(MxK,f16) @ W(NxK,f16)^T + bias ; 4 N-tiles/wave, software-pipelined K -------
__global__ __launch_bounds__(256) void k_gemm(const _Float16* __restrict__ A, int lda,
                                              const _Float16* __restrict__ W, int ldw,
                                              const float* __restrict__ bias0,
                                              const float* __restrict__ bias1,
                                              float* __restrict__ C, _Float16* __restrict__ C16,
                                              int ldc, int M, int N, int K) {
  int lane = threadIdx.x & 31, wave = threadIdx.x >> 5;
  int ngrp = N >> 6;
  int grp = blockIdx.x * 8 + wave;
  if (grp >= (M >> 4) * ngrp) return;              // wave-uniform
  int m0 = (grp / ngrp) * 16, n0 = (grp % ngrp) * 64;
  v8f zero = {0.f, 0.f, 0.f, 0.f, 0.f, 0.f, 0.f, 0.f};
  v8f acc[4] = {zero, zero, zero, zero};
  v16h af = frag_a_f16(A, lda, m0, 0, lane);
  v16h bf[4];
#pragma unroll
  for (int q = 0; q < 4; ++q) bf[q] = frag_b_f16(W, ldw, n0 + q * 16, 0, lane);
  for (int k0 = 0; k0 < K; k0 += 32) {
    v16h ac = af, b0 = bf[0], b1 = bf[1], b2 = bf[2], b3 = bf[3];
    int kn = k0 + 32;
    if (kn < K) {                                  // preload next k-step (overlaps WMMA)
      af = frag_a_f16(A, lda, m0, kn, lane);
#pragma unroll
      for (int q = 0; q < 4; ++q) bf[q] = frag_b_f16(W, ldw, n0 + q * 16, kn, lane);
    }
    acc[0] = wmma16(ac, b0, acc[0]); acc[1] = wmma16(ac, b1, acc[1]);
    acc[2] = wmma16(ac, b2, acc[2]); acc[3] = wmma16(ac, b3, acc[3]);
  }
  int hi = lane >> 4;
#pragma unroll
  for (int q = 0; q < 4; ++q) {
    int n = n0 + q * 16 + (lane & 15);
    float badd = 0.f;
    if (bias0) badd += bias0[n];
    if (bias1) badd += bias1[n];
#pragma unroll
    for (int r = 0; r < 8; ++r) {
      float v = acc[q][r] + badd;
      int row = m0 + r + hi * 8;
      if (C) C[row * ldc + n] = v;
      if (C16) C16[row * ldc + n] = (_Float16)v;
    }
  }
}

// ---------------- persistent biLSTM encoder (one WG per direction; Whh staged in LDS) ----------------
__global__ __launch_bounds__(256) void k_encoder(const float* __restrict__ xg_f,
                                                 const float* __restrict__ xg_b,
                                                 const _Float16* __restrict__ Whh16_f,
                                                 const _Float16* __restrict__ Whh16_b,
                                                 const int* __restrict__ sent_lens,
                                                 float* __restrict__ hs, _Float16* __restrict__ hs16,
                                                 float* __restrict__ h0, float* __restrict__ c0,
                                                 float* __restrict__ out) {
  extern __shared__ char smem[];
  _Float16* h16 = (_Float16*)smem;                 // 64x128 f16
  float* cm = (float*)(smem + 64 * 128 * 2);       // 64x128 f32
  float* h32 = cm + 64 * 128;                      // 64x128 f32
  _Float16* whh = (_Float16*)(h32 + 64 * 128);     // 512x128 f16 (128KB) staged weights
  int dir = blockIdx.x;
  const float* xg = dir ? xg_b : xg_f;
  const _Float16* Wg = dir ? Whh16_b : Whh16_f;
  int tid = threadIdx.x, lane = tid & 31, wave = tid >> 5;
  for (int i = tid; i < 64 * 128; i += 256) { h16[i] = (_Float16)0.f; cm[i] = 0.f; h32[i] = 0.f; }
  for (int i = tid; i < 512 * 128 / 8; i += 256) ((h8*)whh)[i] = ((const h8*)Wg)[i];
  __syncthreads();
  for (int step = 0; step < 100; ++step) {
    int s = dir ? (99 - step) : step;
    for (int j = 0; j < 4; ++j) {                   // 8 waves * 4 groups = 32 (4 mt x 8 ut)
      int g = wave * 4 + j, mt = g >> 3, ut = g & 7;
      v8f zero = {0.f, 0.f, 0.f, 0.f, 0.f, 0.f, 0.f, 0.f};
      v8f a0 = zero, a1 = zero, a2 = zero, a3 = zero;
      v16h af = frag_a_f16(h16, 128, mt * 16, 0, lane);
      v16h w0 = frag_b_f16(whh, 128, 0 * 128 + ut * 16, 0, lane);
      v16h w1 = frag_b_f16(whh, 128, 1 * 128 + ut * 16, 0, lane);
      v16h w2 = frag_b_f16(whh, 128, 2 * 128 + ut * 16, 0, lane);
      v16h w3 = frag_b_f16(whh, 128, 3 * 128 + ut * 16, 0, lane);
      for (int k0 = 0; k0 < 128; k0 += 32) {
        v16h ac = af, c0_ = w0, c1_ = w1, c2_ = w2, c3_ = w3;
        int kn = k0 + 32;
        if (kn < 128) {
          af = frag_a_f16(h16, 128, mt * 16, kn, lane);
          w0 = frag_b_f16(whh, 128, 0 * 128 + ut * 16, kn, lane);
          w1 = frag_b_f16(whh, 128, 1 * 128 + ut * 16, kn, lane);
          w2 = frag_b_f16(whh, 128, 2 * 128 + ut * 16, kn, lane);
          w3 = frag_b_f16(whh, 128, 3 * 128 + ut * 16, kn, lane);
        }
        a0 = wmma16(ac, c0_, a0); a1 = wmma16(ac, c1_, a1);
        a2 = wmma16(ac, c2_, a2); a3 = wmma16(ac, c3_, a3);
      }
      int hi = lane >> 4, u = ut * 16 + (lane & 15);
#pragma unroll
      for (int r = 0; r < 8; ++r) {
        int b = mt * 16 + r + hi * 8, row = b * 100 + s;
        float gi = a0[r] + xg[row * 512 + u];
        float gf = a1[r] + xg[row * 512 + 128 + u];
        float gg = a2[r] + xg[row * 512 + 256 + u];
        float go = a3[r] + xg[row * 512 + 384 + u];
        float co = cm[b * 128 + u], ho = h32[b * 128 + u];
        float cn = sigf(gf) * co + sigf(gi) * tanhf(gg);
        float hn = sigf(go) * tanhf(cn);
        bool vt = s < sent_lens[b];
        cm[b * 128 + u] = vt ? cn : co;
        h32[b * 128 + u] = vt ? hn : ho;
        float henc = vt ? hn : 0.f;
        hs[row * 256 + dir * 128 + u] = henc;
        hs16[row * 256 + dir * 128 + u] = (_Float16)henc;
      }
    }
    __syncthreads();
    for (int i = tid; i < 64 * 128; i += 256) h16[i] = (_Float16)h32[i];
    __syncthreads();
  }
  for (int i = tid; i < 64 * 128; i += 256) {
    int b = i >> 7, u = i & 127;
    h0[b * 256 + dir * 128 + u] = h32[i];
    c0[b * 256 + dir * 128 + u] = cm[i];
    out[64 + b * 256 + dir * 128 + u] = h32[i];     // h_enc_final
  }
}

// ---------------- persistent attention decoder (one WG, 16 waves) ----------------
__global__ __launch_bounds__(512) void k_decoder(
    const _Float16* __restrict__ Wd16,              // 1024 x 960 = [Wih_d | Whh_d]
    const float* __restrict__ bih_d, const float* __restrict__ bhh_d,
    const _Float16* __restrict__ Watt16,            // 256 x 512
    const float* __restrict__ prod_emb, const float* __restrict__ field_emb,
    const int* __restrict__ prev_ids, const int* __restrict__ frontier_ids,
    const int* __restrict__ parent_t, const int* __restrict__ sent_lens,
    const float* __restrict__ h0, const float* __restrict__ c0,
    const float* __restrict__ hs, const float* __restrict__ e_att,
    float* __restrict__ buf, float* __restrict__ satt, _Float16* __restrict__ satt16) {
  extern __shared__ char smem[];
  _Float16* z = (_Float16*)smem;                    // 64 x 960 f16: [prod|s_prev|field|parent|h]
  float* cm = (float*)(smem + 64 * 960 * 2);        // 64 x 256 f32
  float* h32 = cm + 64 * 256;                       // 64 x 256 f32
  int tid = threadIdx.x, lane = tid & 31, wave = tid >> 5;
  for (int i = tid; i < 64 * 960; i += 512) z[i] = (_Float16)0.f;
  for (int i = tid; i < 64 * 256; i += 512) { h32[i] = h0[i]; cm[i] = c0[i]; }
  __syncthreads();
  for (int i = tid; i < 64 * 256; i += 512) { int b = i >> 8, u = i & 255; z[b * 960 + 704 + u] = (_Float16)h32[i]; }
  __syncthreads();
  for (int t = 0; t < 96; ++t) {
    // GEMM1 (pipelined): gates = [inp|h](64x960) @ Wd16^T + bias ; fused LSTM cell
    for (int j = 0; j < 4; ++j) {                   // 16 waves * 4 groups = 64 (4 mt x 16 ut)
      int g = wave * 4 + j, mt = g >> 4, ut = g & 15;
      int hi = lane >> 4, u = ut * 16 + (lane & 15);
      v8f zero = {0.f, 0.f, 0.f, 0.f, 0.f, 0.f, 0.f, 0.f};
      v8f a0 = zero, a1 = zero, a2 = zero, a3 = zero;
      v16h af = frag_a_f16(z, 960, mt * 16, 0, lane);
      v16h w0 = frag_b_f16(Wd16, 960, 0 * 256 + ut * 16, 0, lane);
      v16h w1 = frag_b_f16(Wd16, 960, 1 * 256 + ut * 16, 0, lane);
      v16h w2 = frag_b_f16(Wd16, 960, 2 * 256 + ut * 16, 0, lane);
      v16h w3 = frag_b_f16(Wd16, 960, 3 * 256 + ut * 16, 0, lane);
      for (int k0 = 0; k0 < 960; k0 += 32) {
        v16h ac = af, c0_ = w0, c1_ = w1, c2_ = w2, c3_ = w3;
        int kn = k0 + 32;
        if (kn < 960) {
          af = frag_a_f16(z, 960, mt * 16, kn, lane);
          w0 = frag_b_f16(Wd16, 960, 0 * 256 + ut * 16, kn, lane);
          w1 = frag_b_f16(Wd16, 960, 1 * 256 + ut * 16, kn, lane);
          w2 = frag_b_f16(Wd16, 960, 2 * 256 + ut * 16, kn, lane);
          w3 = frag_b_f16(Wd16, 960, 3 * 256 + ut * 16, kn, lane);
        }
        a0 = wmma16(ac, c0_, a0); a1 = wmma16(ac, c1_, a1);
        a2 = wmma16(ac, c2_, a2); a3 = wmma16(ac, c3_, a3);
      }
#pragma unroll
      for (int r = 0; r < 8; ++r) {
        int b = mt * 16 + r + hi * 8;
        float gi = a0[r] + bih_d[u] + bhh_d[u];
        float gf = a1[r] + bih_d[256 + u] + bhh_d[256 + u];
        float gg = a2[r] + bih_d[512 + u] + bhh_d[512 + u];
        float go = a3[r] + bih_d[768 + u] + bhh_d[768 + u];
        float cn = sigf(gf) * cm[b * 256 + u] + sigf(gi) * tanhf(gg);
        float hn = sigf(go) * tanhf(cn);
        cm[b * 256 + u] = cn; h32[b * 256 + u] = hn;
        buf[(t * 64 + b) * 256 + u] = hn;
      }
    }
    __threadfence();                                // buf visible for parent gathers
    __syncthreads();
    for (int i = tid; i < 64 * 256; i += 512) { int b = i >> 8, u = i & 255; z[b * 960 + 704 + u] = (_Float16)h32[i]; }
    __syncthreads();
    // attention scores aw (float4 dots; aw f32 aliased over dead z cols 0..199)
    for (int p = tid; p < 6400; p += 512) {
      int b = p / 100, s = p % 100;
      const float4* ea = (const float4*)&e_att[(b * 100 + s) * 256];
      const float4* hv = (const float4*)&h32[b * 256];
      float d = 0.f;
      for (int k = 0; k < 64; ++k) {
        float4 a = ea[k], h = hv[k];
        d += a.x * h.x + a.y * h.y + a.z * h.z + a.w * h.w;
      }
      ((float*)(z + b * 960))[s] = d;
    }
    __syncthreads();
    if (tid < 64) {                                 // masked softmax per batch row
      float* aw = (float*)(z + tid * 960);
      int len = sent_lens[tid];
      float mx = -1e30f;
      for (int s = 0; s < len; ++s) mx = fmaxf(mx, aw[s]);
      float sm = 0.f;
      for (int s = 0; s < len; ++s) { float e = expf(aw[s] - mx); aw[s] = e; sm += e; }
      float inv = 1.f / sm;
      for (int s = 0; s < len; ++s) aw[s] *= inv;
      for (int s = len; s < 100; ++s) aw[s] = 0.f;
    }
    __syncthreads();
    for (int i = tid; i < 64 * 64; i += 512) {      // ctx -> z[448:704], 4 units per thread
      int b = i >> 6, u4 = (i & 63) * 4;
      const float* aw = (const float*)(z + b * 960);
      float c0x = 0.f, c1x = 0.f, c2x = 0.f, c3x = 0.f;
      for (int s = 0; s < 100; ++s) {
        float a = aw[s];
        float4 hv = *(const float4*)&hs[(b * 100 + s) * 256 + u4];
        c0x += a * hv.x; c1x += a * hv.y; c2x += a * hv.z; c3x += a * hv.w;
      }
      h4 pk = {(_Float16)c0x, (_Float16)c1x, (_Float16)c2x, (_Float16)c3x};
      *(h4*)&z[b * 960 + 448 + u4] = pk;
    }
    __syncthreads();
    // GEMM2 (pipelined): s = tanh([ctx|h](64x512) @ W_att_vec^T(256x512))
    for (int j = 0; j < 4; ++j) {
      int g = wave * 4 + j, mt = g >> 4, nt = g & 15;
      int hi = lane >> 4, n = nt * 16 + (lane & 15);
      v8f acc = {0.f, 0.f, 0.f, 0.f, 0.f, 0.f, 0.f, 0.f};
      v16h af = frag_a_f16(z + 448, 960, mt * 16, 0, lane);
      v16h wf = frag_b_f16(Watt16, 512, nt * 16, 0, lane);
      for (int k0 = 0; k0 < 512; k0 += 32) {
        v16h ac = af, bc = wf;
        int kn = k0 + 32;
        if (kn < 512) {
          af = frag_a_f16(z + 448, 960, mt * 16, kn, lane);
          wf = frag_b_f16(Watt16, 512, nt * 16, kn, lane);
        }
        acc = wmma16(ac, bc, acc);
      }
#pragma unroll
      for (int r = 0; r < 8; ++r) {
        int b = mt * 16 + r + hi * 8;
        float sv = tanhf(acc[r]);
        satt[(t * 64 + b) * 256 + n] = sv;
        satt16[(t * 64 + b) * 256 + n] = (_Float16)sv;
        z[b * 960 + 128 + n] = (_Float16)sv;        // s_prev for next step
      }
    }
    __syncthreads();
    if (t < 95) {                                   // rebuild inp cols for step t+1
      for (int i = tid; i < 64 * 448; i += 512) {
        int b = i / 448, c = i % 448, tt = t + 1;
        if (c < 128) {
          z[b * 960 + c] = (_Float16)prod_emb[prev_ids[b * 96 + tt] * 128 + c];
        } else if (c < 192) {
          z[b * 960 + 384 + (c - 128)] = (_Float16)field_emb[frontier_ids[b * 96 + tt] * 64 + (c - 128)];
        } else {
          int pt = parent_t[b * 96 + tt];
          z[b * 960 + 448 + (c - 192)] = (_Float16)buf[(pt * 64 + b) * 256 + (c - 192)];
        }
      }
    }
    __syncthreads();
  }
}

// ------- fused logits -> online softmax -> gather (A-fragments hoisted out of N loop) -------
__global__ __launch_bounds__(256) void k_vocab(const _Float16* __restrict__ R,
                                               const _Float16* __restrict__ Wv,
                                               const int* __restrict__ ids,
                                               float* __restrict__ out, int V, int Vpad) {
  __shared__ float redm[16 * 256];
  __shared__ float reds[16 * 256];
  __shared__ float redt[16 * 256];
  int tid = threadIdx.x, lane = tid & 31, wave = tid >> 5;
  int r0 = blockIdx.x * 16, hi = lane >> 4;
  float mrun[8], srun[8], tval[8]; int tgt[8];
#pragma unroll
  for (int r = 0; r < 8; ++r) {
    mrun[r] = -1e30f; srun[r] = 0.f; tval[r] = -1e30f;
    int row = r0 + r + hi * 8;
    tgt[r] = ids[(row & 63) * 96 + (row >> 6)];     // ids[b][t], row = t*64+b
  }
  v16h af0 = frag_a_f16(R, 128, r0, 0, lane);       // whole A row-tile lives in registers
  v16h af1 = frag_a_f16(R, 128, r0, 32, lane);
  v16h af2 = frag_a_f16(R, 128, r0, 64, lane);
  v16h af3 = frag_a_f16(R, 128, r0, 96, lane);
  int ntiles = Vpad >> 4;
  for (int nt = wave; nt < ntiles; nt += 8) {
    int ncol = nt * 16 + (lane & 15);
    v8f acc = {0.f, 0.f, 0.f, 0.f, 0.f, 0.f, 0.f, 0.f};
    acc = wmma16(af0, frag_b_f16(Wv, 128, nt * 16, 0, lane), acc);
    acc = wmma16(af1, frag_b_f16(Wv, 128, nt * 16, 32, lane), acc);
    acc = wmma16(af2, frag_b_f16(Wv, 128, nt * 16, 64, lane), acc);
    acc = wmma16(af3, frag_b_f16(Wv, 128, nt * 16, 96, lane), acc);
    if (ncol + 128 < V) __builtin_prefetch(&Wv[(ncol + 128) * 128], 0, 3);
    if (ncol < V) {
#pragma unroll
      for (int r = 0; r < 8; ++r) {
        float l = acc[r];
        float nm = fmaxf(mrun[r], l);
        srun[r] = srun[r] * expf(mrun[r] - nm) + expf(l - nm);
        mrun[r] = nm;
        if (ncol == tgt[r]) tval[r] = l;
      }
    }
  }
#pragma unroll
  for (int m = 0; m < 16; ++m) {
    bool mine = (m >> 3) == hi;
    int r = m & 7;
    redm[m * 256 + tid] = mine ? mrun[r] : -1e30f;
    reds[m * 256 + tid] = mine ? srun[r] : 0.f;
    redt[m * 256 + tid] = mine ? tval[r] : -1e30f;
  }
  __syncthreads();
  if (tid < 16) {
    float M = -1e30f, S = 0.f, T = -1e30f;
    for (int i = 0; i < 256; ++i) M = fmaxf(M, redm[tid * 256 + i]);
    for (int i = 0; i < 256; ++i) {
      S += reds[tid * 256 + i] * expf(redm[tid * 256 + i] - M);
      T = fmaxf(T, redt[tid * 256 + i]);
    }
    out[r0 + tid] = expf(T - M) / S;
  }
}

// ---------------- copy-pointer scores + masked softmax + weighted sum ----------------
__global__ __launch_bounds__(256) void k_copy(const _Float16* __restrict__ satt16,
                                              const _Float16* __restrict__ hW16,
                                              const float* __restrict__ ict,
                                              const int* __restrict__ sent_lens,
                                              float* __restrict__ out) {
  __shared__ float sc[96 * 112];
  int b = blockIdx.x, tid = threadIdx.x, lane = tid & 31, wave = tid >> 5;
  int hi = lane >> 4;
  for (int tile = wave; tile < 42; tile += 8) {     // 6 t-tiles x 7 s-tiles
    int mt = tile / 7, nt = tile % 7;
    int ncol = nt * 16 + (lane & 15);
    int srow = (ncol < 100) ? ncol : 0;             // clamp; cols >=100 masked later
    v8f acc = {0.f, 0.f, 0.f, 0.f, 0.f, 0.f, 0.f, 0.f};
    for (int k0 = 0; k0 < 256; k0 += 32) {
      v16h af = frag_a_f16(satt16 + b * 256, 64 * 256, mt * 16, k0, lane);
      const _Float16* p = hW16 + (b * 100 + srow) * 256 + k0 + hi * 16;
      v16h bf = cat8(*(const h8*)p, *(const h8*)(p + 8));
      acc = wmma16(af, bf, acc);
    }
#pragma unroll
    for (int r = 0; r < 8; ++r) sc[(mt * 16 + r + hi * 8) * 112 + ncol] = acc[r];
  }
  __syncthreads();
  if (tid < 96) {
    int t = tid, len = sent_lens[b];
    float mx = -1e30f;
    for (int s = 0; s < len; ++s) mx = fmaxf(mx, sc[t * 112 + s]);
    float sm = 0.f;
    for (int s = 0; s < len; ++s) sm += expf(sc[t * 112 + s] - mx);
    float inv = 1.f / sm, acc = 0.f;
    for (int s = 0; s < len; ++s) acc += expf(sc[t * 112 + s] - mx) * inv * ict[(b * 96 + t) * 100 + s];
    out[t * 64 + b] = acc;
  }
}

// ---------------- p_gen + combine + log + reduce over time ----------------
__global__ __launch_bounds__(128) void k_final(const float* __restrict__ satt,
                                               const float* __restrict__ Wgen,
                                               const float* __restrict__ bgen,
                                               const float* __restrict__ tga,
                                               const float* __restrict__ tgg,
                                               const float* __restrict__ tgc,
                                               const float* __restrict__ ma,
                                               const float* __restrict__ mg,
                                               const float* __restrict__ mc,
                                               float* __restrict__ out) {
  __shared__ float red[128];
  int b = blockIdx.x, t = threadIdx.x;
  float lp = 0.f;
  if (t < 96) {
    float d = bgen[0];
    const float4* sa = (const float4*)&satt[(t * 64 + b) * 256];
    const float4* wg = (const float4*)Wgen;
    for (int k = 0; k < 64; ++k) {
      float4 a = sa[k], w = wg[k];
      d += a.x * w.x + a.y * w.y + a.z * w.z + a.w * w.w;
    }
    float pg = 1.f / (1.f + expf(-d));
    float a = ma[b * 96 + t], g = mg[b * 96 + t], c = mc[b * 96 + t];
    float ap = tga[t * 64 + b] * a + pg * tgg[t * 64 + b] * g + (1.f - pg) * tgc[t * 64 + b] * c;
    bool pad = (a + g + c) == 0.f;
    lp = pad ? 0.f : logf(ap);
  }
  red[t] = lp;
  __syncthreads();
  for (int off = 64; off > 0; off >>= 1) {
    if (t < off) red[t] += red[t + off];
    __syncthreads();
  }
  if (t == 0) out[b] = red[0];
}

// ---------------- launch ----------------
extern "C" void kernel_launch(void* const* d_in, const int* in_sizes, int n_in,
                              void* d_out, int out_size, void* d_ws, size_t ws_size,
                              hipStream_t stream) {
  (void)in_sizes; (void)n_in; (void)out_size; (void)ws_size;
  const int*   src_tokens      = (const int*)d_in[0];
  const int*   sent_lens       = (const int*)d_in[1];
  const int*   prev_action_ids = (const int*)d_in[2];
  const int*   parent_t        = (const int*)d_in[3];
  const int*   frontier_ids    = (const int*)d_in[4];
  const int*   applyconstr_ids = (const int*)d_in[5];
  const int*   gentok_ids      = (const int*)d_in[6];
  const float* is_applyconstr  = (const float*)d_in[7];
  const float* is_gentoken     = (const float*)d_in[8];
  const float* is_copy         = (const float*)d_in[9];
  const float* is_copy_tok     = (const float*)d_in[10];
  const float* src_emb   = (const float*)d_in[11];
  const float* prod_emb  = (const float*)d_in[12];
  const float* prim_emb  = (const float*)d_in[13];
  const float* field_emb = (const float*)d_in[14];
  const float* Wih_f = (const float*)d_in[15]; const float* Whh_f = (const float*)d_in[16];
  const float* bih_f = (const float*)d_in[17]; const float* bhh_f = (const float*)d_in[18];
  const float* Wih_b = (const float*)d_in[19]; const float* Whh_b = (const float*)d_in[20];
  const float* bih_b = (const float*)d_in[21]; const float* bhh_b = (const float*)d_in[22];
  const float* Wih_d = (const float*)d_in[23]; const float* Whh_d = (const float*)d_in[24];
  const float* bih_d = (const float*)d_in[25]; const float* bhh_d = (const float*)d_in[26];
  const float* W_lin_attn = (const float*)d_in[27];
  const float* W_att_vec  = (const float*)d_in[28];
  const float* W_ptr      = (const float*)d_in[29];
  const float* W_a2e      = (const float*)d_in[30];
  const float* W_gen      = (const float*)d_in[31];
  const float* b_gen      = (const float*)d_in[32];
  float* out = (float*)d_out;

  // -------- workspace carve (f32 region then f16 region) --------
  float* ws = (float*)d_ws;
  float* xg_f  = ws; ws += 6400 * 512;
  float* xg_b  = ws; ws += 6400 * 512;
  float* hs    = ws; ws += 6400 * 256;       // encodings (premasked)
  float* e_att = ws; ws += 6400 * 256;       // encodings @ W_lin_attn^T
  float* h0    = ws; ws += 64 * 256;
  float* c0    = ws; ws += 64 * 256;
  float* buf   = ws; ws += 96 * 64 * 256;    // decoder states
  float* satt  = ws; ws += 96 * 64 * 256;    // attentional vectors (f32)
  float* tgen  = ws; ws += 6144;
  float* tapply= ws; ws += 6144;
  float* tcopy = ws; ws += 6144;
  _Float16* hp = (_Float16*)ws;
  _Float16* emb16   = hp; hp += 64 * 100 * 128;
  _Float16* hs16    = hp; hp += 6400 * 256;
  _Float16* hW16    = hp; hp += 6400 * 256;  // encodings @ W_ptr^T (f16 only)
  _Float16* satt16  = hp; hp += 6144 * 256;
  _Float16* rdout16 = hp; hp += 6144 * 128;
  _Float16* Wihf16  = hp; hp += 512 * 128;
  _Float16* Wihb16  = hp; hp += 512 * 128;
  _Float16* Whhf16  = hp; hp += 512 * 128;
  _Float16* Whhb16  = hp; hp += 512 * 128;
  _Float16* Wlin16  = hp; hp += 256 * 256;
  _Float16* Wptr16  = hp; hp += 256 * 256;
  _Float16* Watt16  = hp; hp += 256 * 512;
  _Float16* Wa2e16  = hp; hp += 128 * 256;
  _Float16* prim16  = hp; hp += 16000 * 128;
  _Float16* prod16  = hp; hp += 208 * 128;
  _Float16* wd16    = hp; hp += 1024 * 960;

  // -------- prep: f16 conversions --------
  auto cvt = [&](const float* s, _Float16* d, int n) {
    k_tof16<<<(n + 255) / 256, 256, 0, stream>>>(s, d, n);
  };
  k_embed<<<(64 * 100 * 128 + 255) / 256, 256, 0, stream>>>(src_tokens, src_emb, emb16);
  cvt(Wih_f, Wihf16, 512 * 128);  cvt(Wih_b, Wihb16, 512 * 128);
  cvt(Whh_f, Whhf16, 512 * 128);  cvt(Whh_b, Whhb16, 512 * 128);
  cvt(W_lin_attn, Wlin16, 256 * 256); cvt(W_ptr, Wptr16, 256 * 256);
  cvt(W_att_vec, Watt16, 256 * 512);  cvt(W_a2e, Wa2e16, 128 * 256);
  cvt(prim_emb, prim16, 16000 * 128);
  k_prod16<<<(208 * 128 + 255) / 256, 256, 0, stream>>>(prod_emb, prod16);
  k_wd16<<<(1024 * 960 + 255) / 256, 256, 0, stream>>>(Wih_d, Whh_d, wd16);

  // -------- pipeline --------
  k_gemm<<<400, 256, 0, stream>>>(emb16, 128, Wihf16, 128, bih_f, bhh_f, xg_f, nullptr, 512, 6400, 512, 128);
  k_gemm<<<400, 256, 0, stream>>>(emb16, 128, Wihb16, 128, bih_b, bhh_b, xg_b, nullptr, 512, 6400, 512, 128);
  k_encoder<<<2, 256, 64 * 128 * 2 + 64 * 128 * 4 * 2 + 512 * 128 * 2, stream>>>(
      xg_f, xg_b, Whhf16, Whhb16, sent_lens, hs, hs16, h0, c0, out);
  k_gemm<<<200, 256, 0, stream>>>(hs16, 256, Wlin16, 256, nullptr, nullptr, e_att, nullptr, 256, 6400, 256, 256);
  k_gemm<<<200, 256, 0, stream>>>(hs16, 256, Wptr16, 256, nullptr, nullptr, nullptr, hW16, 256, 6400, 256, 256);
  k_decoder<<<1, 512, 64 * 960 * 2 + 64 * 256 * 4 * 2, stream>>>(
      wd16, bih_d, bhh_d, Watt16, prod_emb, field_emb, prev_action_ids, frontier_ids, parent_t,
      sent_lens, h0, c0, hs, e_att, buf, satt, satt16);
  k_gemm<<<96, 256, 0, stream>>>(satt16, 256, Wa2e16, 256, nullptr, nullptr, nullptr, rdout16, 128, 6144, 128, 256);
  k_vocab<<<384, 256, 0, stream>>>(rdout16, prim16, gentok_ids, tgen, 16000, 16000);
  k_vocab<<<384, 256, 0, stream>>>(rdout16, prod16, applyconstr_ids, tapply, 201, 208);
  k_copy<<<64, 256, 0, stream>>>(satt16, hW16, is_copy_tok, sent_lens, tcopy);
  k_final<<<64, 128, 0, stream>>>(satt, W_gen, b_gen, tapply, tgen, tcopy, is_applyconstr,
                                  is_gentoken, is_copy, out);
}
